// RGCNLayer_5952824672811
// MI455X (gfx1250) — matrix operand order repaired and checked by reading, not compiled, not run on hardware
//
#include <hip/hip_runtime.h>
#include <hip/hip_bf16.h>

typedef __bf16 bf16_t;
typedef __attribute__((ext_vector_type(16))) __bf16 v16bf;
typedef __attribute__((ext_vector_type(8)))  __bf16 v8bf;
typedef __attribute__((ext_vector_type(4)))  __bf16 v4bf;
typedef __attribute__((ext_vector_type(8)))  float  v8f;
typedef __attribute__((ext_vector_type(4)))  float  v4f;

#define FEAT  128
#define NRELS 8
#define APAD  136   // row pitch in elements: 136*2B = 272B, multiple of 16B

union FragAB { v16bf v; v8bf h[2]; };

// ---------------------------------------------------------------------------
// Phase 1: h_all[n,r,:] = h[n,:] @ W[r]   (WMMA bf16 -> f32 accum)
//          gate_all[n,r] = h[n,:] . gw[r]   (logit, sigmoid applied later)
// Block = 128 threads = 4 waves. Each wave owns one 16-row tile, all 128 cols.
// grid = (ceil(rowTiles/4), NRELS)
// ---------------------------------------------------------------------------
template <typename TOUT>
__global__ void __launch_bounds__(128)
rgcn_transform(const float* __restrict__ h,
               const float* __restrict__ weight,
               const float* __restrict__ gate_weight,
               TOUT*  __restrict__ h_all,
               float* __restrict__ gate_all,
               int nNodes, int rowTiles) {
  __shared__ __attribute__((aligned(16))) bf16_t sWt[FEAT][APAD];   // transposed: [f][d]
  __shared__ __attribute__((aligned(16))) bf16_t sA[4][16][APAD];   // per-wave: [m][k]
  __shared__ __attribute__((aligned(16))) float  sGW[FEAT];

  const int r    = blockIdx.y;
  const int tid  = threadIdx.x;
  const int w    = tid >> 5;
  const int lane = tid & 31;
  const int m    = lane & 15;
  const int hi   = lane >> 4;

  // Cooperative load of W_r (coalesced float4), transposed into LDS as bf16.
  const float* Wr = weight + (size_t)r * FEAT * FEAT;
  #pragma unroll
  for (int i = 0; i < (FEAT * FEAT) / (128 * 4); ++i) {
    int idx = (tid + i * 128) * 4;        // 4 consecutive f in row d
    int d = idx >> 7;
    int f = idx & 127;
    v4f vv = *(const v4f*)&Wr[idx];
    sWt[f + 0][d] = (bf16_t)vv[0];
    sWt[f + 1][d] = (bf16_t)vv[1];
    sWt[f + 2][d] = (bf16_t)vv[2];
    sWt[f + 3][d] = (bf16_t)vv[3];
  }
  sGW[tid] = gate_weight[r * FEAT + tid];   // blockDim == FEAT

  // This wave's 16 rows of h -> LDS bf16 (float4 global loads, b64 LDS stores).
  const int tile    = blockIdx.x * 4 + w;
  const int rowBase = tile * 16;
  if (tile < rowTiles) {
    #pragma unroll
    for (int i = 0; i < (16 * FEAT) / (32 * 4); ++i) {
      int idx  = (lane + i * 32) * 4;     // 4 consecutive cc in row rr
      int rr   = idx >> 7;
      int cc   = idx & 127;
      int gRow = rowBase + rr;
      v4f vv = {0.f, 0.f, 0.f, 0.f};
      if (gRow < nNodes) vv = *(const v4f*)&h[(size_t)gRow * FEAT + cc];
      v4bf b;
      b[0] = (bf16_t)vv[0]; b[1] = (bf16_t)vv[1];
      b[2] = (bf16_t)vv[2]; b[3] = (bf16_t)vv[3];
      *(v4bf*)&sA[w][rr][cc] = b;
    }
  }
  __syncthreads();
  if (tile >= rowTiles) return;

  // A fragments: ISA 16-bit 16x32 A layout.
  // lane (m, hi): K = c*32 + {8*hi + 0..7} and {16 + 8*hi + 0..7}  (contiguous runs)
  FragAB a[4];
  #pragma unroll
  for (int c = 0; c < 4; ++c) {
    a[c].h[0] = *(const v8bf*)&sA[w][m][c * 32 + 8 * hi];
    a[c].h[1] = *(const v8bf*)&sA[w][m][c * 32 + 16 + 8 * hi];
  }

  // One base pointer; all 64 stores use compile-time offsets (i*4096B + t*64B f32).
  const size_t rowStride = (size_t)NRELS * FEAT;                 // elements
  TOUT* outBase = h_all + ((size_t)(rowBase + 8 * hi) * NRELS + r) * FEAT + m;
  const bool fullTile = (rowBase + 16) <= nNodes;

  // 8 column tiles of 16, K = 4 chunks of 32.
  #pragma unroll
  for (int t = 0; t < 8; ++t) {
    v8f acc = {0.f, 0.f, 0.f, 0.f, 0.f, 0.f, 0.f, 0.f};
    #pragma unroll
    for (int c = 0; c < 4; ++c) {
      // B layout: lane holds column N = m, K = c*32 + {8*hi + 0..7} and {16 + 8*hi + 0..7} (matches A)
      FragAB b;
      b.h[0] = *(const v8bf*)&sWt[t * 16 + m][c * 32 + 8 * hi];
      b.h[1] = *(const v8bf*)&sWt[t * 16 + m][c * 32 + 16 + 8 * hi];
      acc = __builtin_amdgcn_wmma_f32_16x16x32_bf16(
          false, a[c].v, false, b.v, (short)0, acc, false, false);
    }
    // C/D layout: VGPR i -> row (i + 8*hi), col = m. Stream out (non-temporal).
    TOUT* p = outBase + t * 16;
    if (fullTile) {
      #pragma unroll
      for (int i = 0; i < 8; ++i)
        __builtin_nontemporal_store((TOUT)acc[i], p + (size_t)i * rowStride);
    } else {
      #pragma unroll
      for (int i = 0; i < 8; ++i)
        if (rowBase + 8 * hi + i < nNodes)
          __builtin_nontemporal_store((TOUT)acc[i], p + (size_t)i * rowStride);
    }
  }

  // Gate logit: lanes split the 128-deep dot in halves, combine via shuffle.
  float s = 0.f;
  const int dbase = hi * 64;
  #pragma unroll
  for (int c8 = 0; c8 < 8; ++c8) {
    v8bf av = *(const v8bf*)&sA[w][m][dbase + c8 * 8];
    v4f g0 = *(const v4f*)&sGW[dbase + c8 * 8];
    v4f g1 = *(const v4f*)&sGW[dbase + c8 * 8 + 4];
    s += (float)av[0] * g0[0] + (float)av[1] * g0[1]
       + (float)av[2] * g0[2] + (float)av[3] * g0[3]
       + (float)av[4] * g1[0] + (float)av[5] * g1[1]
       + (float)av[6] * g1[2] + (float)av[7] * g1[3];
  }
  s += __shfl_xor(s, 16, 32);
  int gRow = rowBase + m;
  if (hi == 0 && gRow < nNodes)
    gate_all[(size_t)gRow * NRELS + r] = s;
}

// ---------------------------------------------------------------------------
// Phase 2: per-edge gather + scale + scatter-add. One 32-lane group per edge.
// ---------------------------------------------------------------------------
__device__ inline void load4(const float* p, int lane, float v[4]) {
  v4f t = __builtin_nontemporal_load((const v4f*)p + lane);
  v[0] = t[0]; v[1] = t[1]; v[2] = t[2]; v[3] = t[3];
}
__device__ inline void load4(const bf16_t* p, int lane, float v[4]) {
  v4bf t = __builtin_nontemporal_load((const v4bf*)p + lane);
  v[0] = (float)t[0]; v[1] = (float)t[1]; v[2] = (float)t[2]; v[3] = (float)t[3];
}

template <typename TIN>
__global__ void __launch_bounds__(256)
rgcn_edge(const TIN*  __restrict__ h_all,
          const float* __restrict__ gate_all,
          const float* __restrict__ norm,
          const int*   __restrict__ src,
          const int*   __restrict__ dst,
          const int*   __restrict__ rel,
          float* __restrict__ out, int nEdges) {
  int gid  = blockIdx.x * 256 + threadIdx.x;
  int e    = gid >> 5;
  int lane = threadIdx.x & 31;
  if (e >= nEdges) return;

  int s = src[e], r = rel[e], d = dst[e];
  float logit = gate_all[(size_t)s * NRELS + r];
  float c = norm[e] * (1.0f / (1.0f + __expf(-logit)));

  const TIN* row = h_all + ((size_t)s * NRELS + r) * FEAT;
  float v[4];
  load4(row, lane, v);

  float* o = out + (size_t)d * FEAT + lane * 4;
  atomicAdd(o + 0, v[0] * c);
  atomicAdd(o + 1, v[1] * c);
  atomicAdd(o + 2, v[2] * c);
  atomicAdd(o + 3, v[3] * c);
}

// ---------------------------------------------------------------------------
__global__ void __launch_bounds__(256) zero_f32(float* p, int n) {
  int i = blockIdx.x * 256 + threadIdx.x;
  if (i < n) p[i] = 0.0f;
}
__global__ void __launch_bounds__(256) relu_f32(float* p, int n) {
  int i = blockIdx.x * 256 + threadIdx.x;
  if (i < n) p[i] = fmaxf(p[i], 0.0f);
}

// ---------------------------------------------------------------------------
extern "C" void kernel_launch(void* const* d_in, const int* in_sizes, int n_in,
                              void* d_out, int out_size, void* d_ws, size_t ws_size,
                              hipStream_t stream) {
  (void)n_in; (void)out_size;
  const float* h    = (const float*)d_in[0];
  const float* W    = (const float*)d_in[1];
  const float* gw   = (const float*)d_in[2];
  const float* norm = (const float*)d_in[3];
  const int*   src  = (const int*)d_in[4];
  const int*   dst  = (const int*)d_in[5];
  const int*   rel  = (const int*)d_in[6];
  float* out = (float*)d_out;

  const int N = in_sizes[0] / FEAT;
  const int E = in_sizes[4];
  const int rowTiles = (N + 15) / 16;

  const size_t nPairs   = (size_t)N * NRELS;
  const size_t needF32  = nPairs * FEAT * sizeof(float) + nPairs * sizeof(float);
  const bool   useF32   = ws_size >= needF32;

  const int outN = N * FEAT;
  zero_f32<<<(outN + 255) / 256, 256, 0, stream>>>(out, outN);

  dim3 tgrid((rowTiles + 3) / 4, NRELS);
  if (useF32) {
    float* h_all = (float*)d_ws;
    float* gate  = h_all + nPairs * FEAT;
    rgcn_transform<float><<<tgrid, 128, 0, stream>>>(h, W, gw, h_all, gate, N, rowTiles);
    rgcn_edge<float><<<(E + 7) / 8, 256, 0, stream>>>(h_all, gate, norm, src, dst, rel, out, E);
  } else {
    bf16_t* h_all = (bf16_t*)d_ws;
    float*  gate  = (float*)(h_all + nPairs * FEAT);
    rgcn_transform<bf16_t><<<tgrid, 128, 0, stream>>>(h, W, gw, h_all, gate, N, rowTiles);
    rgcn_edge<bf16_t><<<(E + 7) / 8, 256, 0, stream>>>(h_all, gate, norm, src, dst, rel, out, E);
  }

  relu_f32<<<(outN + 255) / 256, 256, 0, stream>>>(out, outN);
}
